// ExpertLayer_80169859548045
// MI455X (gfx1250) — matrix operand order repaired and checked by reading, not compiled, not run on hardware
//
#include <hip/hip_runtime.h>
#include <hip/hip_bf16.h>

#define D_DIM 1024
#define E_DIM 8
#define NTOK  8192   // B*S = 4*2048
#define MT    128
#define NT    128
#define KT    64
#define LDA   72     // padded LDS row stride in bf16 (144B = 36 dwords -> conflict-free)
#define NIT   (E_DIM * (D_DIM / KT))   // 128 pipeline iterations

typedef __attribute__((ext_vector_type(8)))  float   v8f;
typedef __attribute__((ext_vector_type(4)))  float   v4f;
typedef __attribute__((ext_vector_type(4)))  int     v4i;
typedef __attribute__((ext_vector_type(8)))  __bf16  v8bf;
typedef __attribute__((ext_vector_type(16))) __bf16  v16bf;

// ---------------- CDNA5 async global->LDS copy helpers ---------------------------
// Builtin prototype (from clang diagnostic): param0 = AS1 int4*, param1 = AS3 int4*.
__device__ __forceinline__ void async_copy_b128(const void* g, void* l) {
#if __has_builtin(__builtin_amdgcn_global_load_async_to_lds_b128)
  __builtin_amdgcn_global_load_async_to_lds_b128(
      (__attribute__((address_space(1))) v4i*)g,
      (__attribute__((address_space(3))) v4i*)l, 0, 0);
#else
  *(v8bf*)l = *(const v8bf*)g;   // synchronous fallback (ds path)
#endif
}
__device__ __forceinline__ void async_wait0() {
#if __has_builtin(__builtin_amdgcn_global_load_async_to_lds_b128)
#if __has_builtin(__builtin_amdgcn_s_wait_asynccnt)
  __builtin_amdgcn_s_wait_asynccnt(0);
#else
  asm volatile("s_wait_asynccnt 0x0" ::: "memory");
#endif
#endif
}

// ---------------- Router: logits -> softmax weights, one wave per token ----------
__global__ __launch_bounds__(256) void moe_router(const float* __restrict__ x,
                                                  const float* __restrict__ Wr,
                                                  const float* __restrict__ br,
                                                  float* __restrict__ wts) {
  const int lane  = threadIdx.x & 31;
  const int wave  = threadIdx.x >> 5;
  const int token = blockIdx.x * 8 + wave;

  float xi[32];
  const float* xr = x + (size_t)token * D_DIM + lane;
#pragma unroll
  for (int i = 0; i < 32; ++i) xi[i] = xr[i * 32];

  float lg[E_DIM];
#pragma unroll
  for (int e = 0; e < E_DIM; ++e) {
    const float* wr = Wr + (size_t)e * D_DIM + lane;
    float s = 0.f;
#pragma unroll
    for (int i = 0; i < 32; ++i) s = fmaf(xi[i], wr[i * 32], s);
#pragma unroll
    for (int off = 16; off > 0; off >>= 1) s += __shfl_xor(s, off, 32);
    lg[e] = s + br[e];
  }
  float m = lg[0];
#pragma unroll
  for (int e = 1; e < E_DIM; ++e) m = fmaxf(m, lg[e]);
  float sum = 0.f;
#pragma unroll
  for (int e = 0; e < E_DIM; ++e) { lg[e] = __expf(lg[e] - m); sum += lg[e]; }
  const float inv = 1.f / sum;
  if (lane == 0) {
    v4f a = { lg[0] * inv, lg[1] * inv, lg[2] * inv, lg[3] * inv };
    v4f b = { lg[4] * inv, lg[5] * inv, lg[6] * inv, lg[7] * inv };
    *(v4f*)(wts + (size_t)token * E_DIM)     = a;
    *(v4f*)(wts + (size_t)token * E_DIM + 4) = b;
  }
}

// ---------------- fp32 -> bf16 bulk conversion (bandwidth-bound, one pass) -------
__global__ __launch_bounds__(256) void conv_bf16(const float* __restrict__ s,
                                                 __bf16* __restrict__ d) {
  const size_t i = ((size_t)blockIdx.x * 256 + threadIdx.x) * 8;
  float4 a0 = *(const float4*)(s + i);
  float4 a1 = *(const float4*)(s + i + 4);
  v8bf o = { (__bf16)a0.x, (__bf16)a0.y, (__bf16)a0.z, (__bf16)a0.w,
             (__bf16)a1.x, (__bf16)a1.y, (__bf16)a1.z, (__bf16)a1.w };
  *(v8bf*)(d + i) = o;
}

// ---------------- Main MoE GEMM, async-staged, ping-pong LDS ---------------------
// accE_e = X @ W_e^T over K; every 16 iters: accF += w[:,e] * accE_e.
__global__ __launch_bounds__(256) void moe_gemm_async(
    const __bf16* __restrict__ xb,   // [NTOK, D] bf16
    const __bf16* __restrict__ wb,   // [E, D, D] bf16
    const float*  __restrict__ be,   // [E, D]
    const float*  __restrict__ wts,  // [NTOK, E]
    float* __restrict__ out) {       // [NTOK, D]
  __shared__ __bf16 sA[2][MT * LDA];
  __shared__ __bf16 sB[2][NT * LDA];
  __shared__ float  sW[MT * E_DIM];
  __shared__ float  sBias[E_DIM * NT];

  const int n0   = blockIdx.x * NT;
  const int m0   = blockIdx.y * MT;
  const int t    = threadIdx.x;
  const int lane = t & 31;
  const int wave = t >> 5;
  const int wm   = wave >> 1;   // 0..3 -> M offset 32*wm
  const int wn   = wave & 1;    // 0..1 -> N offset 64*wn
  const int radd = (lane >> 4) * 8;

  for (int i = t; i < MT * E_DIM; i += 256)
    sW[i] = wts[(size_t)(m0 + (i >> 3)) * E_DIM + (i & 7)];
  for (int i = t; i < E_DIM * NT; i += 256)
    sBias[i] = be[(size_t)(i >> 7) * D_DIM + n0 + (i & 127)];

  const int lrow  = t >> 1;        // 0..127: tile row this thread stages
  const int cbase = (t & 1) * 4;   // first 16B chunk within the row

  // stage iteration 'it' into buffer 'buf' (pure async copies, no VALU)
  auto stage = [&](int it, int buf) {
    const int e  = it >> 4;
    const int k0 = (it & 15) * KT;
    const __bf16* srcA = xb + (size_t)(m0 + lrow) * D_DIM + k0 + cbase * 8;
    __bf16* dstA = &sA[buf][lrow * LDA + cbase * 8];
#pragma unroll
    for (int c = 0; c < 4; ++c) async_copy_b128(srcA + c * 8, dstA + c * 8);
    const __bf16* srcB = wb + ((size_t)e * D_DIM + n0 + lrow) * D_DIM + k0 + cbase * 8;
    __bf16* dstB = &sB[buf][lrow * LDA + cbase * 8];
#pragma unroll
    for (int c = 0; c < 4; ++c) async_copy_b128(srcB + c * 8, dstB + c * 8);
  };

  v8f accF[2][4] = {};   // final routing-weighted sum
  v8f accE[2][4] = {};   // current expert partial

  stage(0, 0);

  for (int it = 0; it < NIT; ++it) {
    async_wait0();
    __syncthreads();
    if (it + 1 < NIT) stage(it + 1, (it + 1) & 1);

    const __bf16* sAb = sA[it & 1];
    const __bf16* sBb = sB[it & 1];
#pragma unroll
    for (int kk = 0; kk < KT; kk += 32) {
      v16bf af[2], bfrag[4];
      const int asel = (lane >> 4) * 8;   // A: lanes16-31 hold K 8-15 / 24-31
      const int bsel = (lane >> 4) * 8;   // B: lanes16-31 hold K 8-15 / 24-31
#pragma unroll
      for (int i = 0; i < 2; ++i) {
        const __bf16* p = &sAb[(wm * 32 + i * 16 + (lane & 15)) * LDA + kk + asel];
        v8bf lo = *(const v8bf*)p;
        v8bf hi = *(const v8bf*)(p + 16);
        af[i] = __builtin_shufflevector(lo, hi, 0,1,2,3,4,5,6,7,8,9,10,11,12,13,14,15);
      }
#pragma unroll
      for (int j = 0; j < 4; ++j) {
        const __bf16* p = &sBb[(wn * 64 + j * 16 + (lane & 15)) * LDA + kk + bsel];
        v8bf lo = *(const v8bf*)p;
        v8bf hi = *(const v8bf*)(p + 16);
        bfrag[j] = __builtin_shufflevector(lo, hi, 0,1,2,3,4,5,6,7,8,9,10,11,12,13,14,15);
      }
#pragma unroll
      for (int i = 0; i < 2; ++i)
#pragma unroll
        for (int j = 0; j < 4; ++j)
          accE[i][j] = __builtin_amdgcn_wmma_f32_16x16x32_bf16(
              false, af[i], false, bfrag[j], (short)0, accE[i][j], false, false);
    }

    // expert finished: fold into routing-weighted accumulator, reset expert acc
    if (((it + 1) & 15) == 0) {
      const int e = it >> 4;
#pragma unroll
      for (int i = 0; i < 2; ++i) {
#pragma unroll
        for (int r = 0; r < 8; ++r) {
          const int row = wm * 32 + i * 16 + r + radd;
          const float w = sW[row * E_DIM + e];
#pragma unroll
          for (int j = 0; j < 4; ++j) {
            accF[i][j][r] = fmaf(w, accE[i][j][r], accF[i][j][r]);
            accE[i][j][r] = 0.f;
          }
        }
      }
    }
  }

  // ---- epilogue: add routing-weighted bias, store fp32 ----
  const int colq = lane & 15;
#pragma unroll
  for (int i = 0; i < 2; ++i) {
#pragma unroll
    for (int j = 0; j < 4; ++j) {
      const int col = wn * 64 + j * 16 + colq;
#pragma unroll
      for (int r = 0; r < 8; ++r) {
        const int row = wm * 32 + i * 16 + r + radd;
        float bias = 0.f;
#pragma unroll
        for (int e = 0; e < E_DIM; ++e)
          bias = fmaf(sW[row * E_DIM + e], sBias[e * NT + col], bias);
        out[(size_t)(m0 + row) * D_DIM + (n0 + col)] = accF[i][j][r] + bias;
      }
    }
  }
}

// ---------------- Fallback (small workspace): fused fp32->bf16 in-loop ----------
__global__ __launch_bounds__(256) void moe_gemm_fallback(
    const float* __restrict__ x, const float* __restrict__ We,
    const float* __restrict__ be, const float* __restrict__ wts,
    float* __restrict__ out) {
  __shared__ __bf16 sA[MT * LDA];
  __shared__ __bf16 sB[NT * LDA];
  __shared__ float  sW[MT * E_DIM];
  __shared__ float  sBias[E_DIM * NT];

  const int n0 = blockIdx.x * NT, m0 = blockIdx.y * MT;
  const int t = threadIdx.x, lane = t & 31, wave = t >> 5;
  const int wm = wave >> 1, wn = wave & 1;

  for (int i = t; i < MT * E_DIM; i += 256)
    sW[i] = wts[(size_t)(m0 + (i >> 3)) * E_DIM + (i & 7)];
  for (int i = t; i < E_DIM * NT; i += 256)
    sBias[i] = be[(size_t)(i >> 7) * D_DIM + n0 + (i & 127)];
  __syncthreads();

  v8f acc[2][4] = {};
  const int lrow = t >> 1, lhalf = t & 1;

  for (int e = 0; e < E_DIM; ++e) {
    const float wscale = sW[lrow * E_DIM + e];
    for (int k0 = 0; k0 < D_DIM; k0 += KT) {
      {
        const float* src = x + (size_t)(m0 + lrow) * D_DIM + k0 + lhalf * 32;
        __bf16* dst = &sA[lrow * LDA + lhalf * 32];
#pragma unroll
        for (int c = 0; c < 4; ++c) {
          float4 a0 = *(const float4*)(src + c * 8);
          float4 a1 = *(const float4*)(src + c * 8 + 4);
          v8bf o = { (__bf16)(a0.x * wscale), (__bf16)(a0.y * wscale),
                     (__bf16)(a0.z * wscale), (__bf16)(a0.w * wscale),
                     (__bf16)(a1.x * wscale), (__bf16)(a1.y * wscale),
                     (__bf16)(a1.z * wscale), (__bf16)(a1.w * wscale) };
          *(v8bf*)(dst + c * 8) = o;
        }
      }
      {
        const float* src = We + ((size_t)e * D_DIM + (n0 + lrow)) * D_DIM + k0 + lhalf * 32;
        __bf16* dst = &sB[lrow * LDA + lhalf * 32];
#pragma unroll
        for (int c = 0; c < 4; ++c) {
          float4 a0 = *(const float4*)(src + c * 8);
          float4 a1 = *(const float4*)(src + c * 8 + 4);
          v8bf o = { (__bf16)a0.x, (__bf16)a0.y, (__bf16)a0.z, (__bf16)a0.w,
                     (__bf16)a1.x, (__bf16)a1.y, (__bf16)a1.z, (__bf16)a1.w };
          *(v8bf*)(dst + c * 8) = o;
        }
      }
      __syncthreads();
#pragma unroll
      for (int kk = 0; kk < KT; kk += 32) {
        v16bf af[2], bfrag[4];
        const int asel = (lane >> 4) * 8, bsel = (lane >> 4) * 8;
#pragma unroll
        for (int i = 0; i < 2; ++i) {
          const __bf16* p = &sA[(wm * 32 + i * 16 + (lane & 15)) * LDA + kk + asel];
          v8bf lo = *(const v8bf*)p;
          v8bf hi = *(const v8bf*)(p + 16);
          af[i] = __builtin_shufflevector(lo, hi, 0,1,2,3,4,5,6,7,8,9,10,11,12,13,14,15);
        }
#pragma unroll
        for (int j = 0; j < 4; ++j) {
          const __bf16* p = &sB[(wn * 64 + j * 16 + (lane & 15)) * LDA + kk + bsel];
          v8bf lo = *(const v8bf*)p;
          v8bf hi = *(const v8bf*)(p + 16);
          bfrag[j] = __builtin_shufflevector(lo, hi, 0,1,2,3,4,5,6,7,8,9,10,11,12,13,14,15);
        }
#pragma unroll
        for (int i = 0; i < 2; ++i)
#pragma unroll
          for (int j = 0; j < 4; ++j)
            acc[i][j] = __builtin_amdgcn_wmma_f32_16x16x32_bf16(
                false, af[i], false, bfrag[j], (short)0, acc[i][j], false, false);
      }
      __syncthreads();
    }
  }

  const int colq = lane & 15, radd = (lane >> 4) * 8;
#pragma unroll
  for (int i = 0; i < 2; ++i)
#pragma unroll
    for (int j = 0; j < 4; ++j) {
      const int col = wn * 64 + j * 16 + colq;
#pragma unroll
      for (int r = 0; r < 8; ++r) {
        const int row = wm * 32 + i * 16 + r + radd;
        float bias = 0.f;
#pragma unroll
        for (int e = 0; e < E_DIM; ++e)
          bias = fmaf(sW[row * E_DIM + e], sBias[e * NT + col], bias);
        out[(size_t)(m0 + row) * D_DIM + (n0 + col)] = acc[i][j][r] + bias;
      }
    }
}

extern "C" void kernel_launch(void* const* d_in, const int* in_sizes, int n_in,
                              void* d_out, int out_size, void* d_ws, size_t ws_size,
                              hipStream_t stream) {
  const float* x  = (const float*)d_in[0];
  const float* We = (const float*)d_in[1];
  const float* be = (const float*)d_in[2];
  const float* Wr = (const float*)d_in[3];
  const float* br = (const float*)d_in[4];
  float* out = (float*)d_out;

  const size_t n_x = (size_t)NTOK * D_DIM;          // 8.4M
  const size_t n_w = (size_t)E_DIM * D_DIM * D_DIM; // 8.4M
  const size_t off_xb = 256 * 1024;                 // after router weights
  const size_t off_wb = off_xb + n_x * sizeof(__bf16);
  const size_t need   = off_wb + n_w * sizeof(__bf16);

  float* wts = (float*)d_ws;   // NTOK*E floats = 256 KB
  moe_router<<<NTOK / 8, 256, 0, stream>>>(x, Wr, br, wts);

  dim3 grid(D_DIM / NT, NTOK / MT);
  if (ws_size >= need) {
    __bf16* xb = (__bf16*)((char*)d_ws + off_xb);
    __bf16* wb = (__bf16*)((char*)d_ws + off_wb);
    conv_bf16<<<(int)(n_x / 2048), 256, 0, stream>>>(x, xb);
    conv_bf16<<<(int)(n_w / 2048), 256, 0, stream>>>(We, wb);
    moe_gemm_async<<<grid, 256, 0, stream>>>(xb, wb, be, wts, out);
  } else {
    moe_gemm_fallback<<<grid, 256, 0, stream>>>(x, We, be, wts, out);
  }
}
